// DualVit_54494545052361
// MI455X (gfx1250) — matrix operand order repaired and checked by reading, not compiled, not run on hardware
//
#include <hip/hip_runtime.h>
#include <hip/hip_bf16.h>

// ---------------------------------------------------------------------------
// Types for CDNA5 WMMA (wave32, 16x16x32 bf16 -> f32)
// ---------------------------------------------------------------------------
typedef __attribute__((ext_vector_type(16))) __bf16 v16bf;
typedef __attribute__((ext_vector_type(8)))  float  v8f;
typedef __attribute__((ext_vector_type(4)))  float  f32x4;

#define TOK   50176      // 16 * 56 * 56  ==  1024 windows * 49 tokens
#define DIMC  384
#define HID   1536
#define NHEAD 12
#define HD    32
#define NWIN  1024
#define QSCALE 0.17677669529663687f   // 32^-0.5

__device__ __forceinline__ v16bf zero_frag16() {
    v16bf z;
    f32x4 zz = {0.f, 0.f, 0.f, 0.f};
    ((f32x4*)&z)[0] = zz;
    ((f32x4*)&z)[1] = zz;
    return z;
}

__device__ __forceinline__ v8f zero_acc() {
    v8f c;
#pragma unroll
    for (int i = 0; i < 8; ++i) c[i] = 0.f;
    return c;
}

// A-fragment (16x32 bf16) from a pointer already positioned at row start + k0:
// lanes 0-15 : halves 0-15 = K +0..15
// lanes 16-31: halves 0-15 = K +16..31
// (caller adds the +16 half-lane offset)
__device__ __forceinline__ v16bf load_a_frag_ptr(const __bf16* p) {
    v16bf r;
    ((f32x4*)&r)[0] = *(const f32x4*)(p);
    ((f32x4*)&r)[1] = *(const f32x4*)(p + 8);
    return r;
}

__device__ __forceinline__ v16bf load_a_frag(const __bf16* A, int row, int ldk,
                                             int k0, int lane) {
    return load_a_frag_ptr(A + (size_t)row * ldk + k0 + ((lane >= 16) ? 16 : 0));
}

// B-fragment from a K-contiguous (transposed) matrix Bt[ncols][ldk]:
// lanes 0-15 : col = n, K = k0+0..15 ; lanes 16-31: col = n, K = k0+16..31
__device__ __forceinline__ v16bf load_b_frag_rowmajor(const __bf16* Bt, int col,
                                                      int ldk, int k0, int lane) {
    const __bf16* p = Bt + (size_t)col * ldk + k0 + ((lane >= 16) ? 16 : 0);
    v16bf r;
    ((f32x4*)&r)[0] = *(const f32x4*)(p);
    ((f32x4*)&r)[1] = *(const f32x4*)(p + 8);
    return r;
}

// B-fragment from packed weights, pointer already at lane's 16 bf16
__device__ __forceinline__ v16bf load_b_frag_ptr(const __bf16* p) {
    v16bf r;
    ((f32x4*)&r)[0] = *(const f32x4*)(p);
    ((f32x4*)&r)[1] = *(const f32x4*)(p + 8);
    return r;
}

__device__ __forceinline__ v8f wmma_bf16(v16bf a, v16bf b, v8f c) {
    return __builtin_amdgcn_wmma_f32_16x16x32_bf16(
        false, a, false, b, (short)0, c, false, false);
}

// ---------------------------------------------------------------------------
// Weight packing: fp32 W[K][ld] (column offset coloff, N columns used) -> bf16
// fragment tiles: tile (nt16, kt) holds 32 lanes * 16 bf16 contiguous,
// tile index = nt16*KT + kt  (so kt-consecutive tiles are 1024 B apart).
// ---------------------------------------------------------------------------
__global__ void packw_kernel(const float* __restrict__ W, int ld, int coloff,
                             int K, int N, __bf16* __restrict__ out) {
    int idx = blockIdx.x * blockDim.x + threadIdx.x;
    int KT = K >> 5, NT = N >> 4;
    if (idx >= KT * NT * 32) return;
    int lane = idx & 31;
    int tile = idx >> 5;          // tile = nt*KT + kt
    int kt = tile % KT;
    int nt = tile / KT;
    int rbase = kt * 32 + ((lane >= 16) ? 16 : 0);
    int c = coloff + nt * 16 + (lane & 15);
    __bf16* o = out + (size_t)idx * 16;
#pragma unroll
    for (int i = 0; i < 16; ++i)
        o[i] = (__bf16)W[(size_t)(rbase + i) * ld + c];
}

// ---------------------------------------------------------------------------
// kv prep: kv fp32 (2,16,12,49,32) -> k bf16 (row-major 49x32 per b,h) and
// vT bf16 (32 x 64, zero padded K) per b,h.
// ---------------------------------------------------------------------------
__global__ void kvprep_kernel(const float* __restrict__ kv,
                              __bf16* __restrict__ kbf,
                              __bf16* __restrict__ vT) {
    int idx = blockIdx.x * blockDim.x + threadIdx.x;
    const int KELEMS = 16 * NHEAD * 49 * HD;      // 301056
    const int VELEMS = 16 * NHEAD * HD * 64;      // 393216
    if (idx < KELEMS) kbf[idx] = (__bf16)kv[idx];
    if (idx < VELEMS) {
        int t  = idx & 63;
        int d  = (idx >> 6) & 31;
        int bh = idx >> 11;
        float val = (t < 49) ? kv[KELEMS + ((size_t)bh * 49 + t) * HD + d] : 0.f;
        vT[idx] = (__bf16)val;
    }
}

// ---------------------------------------------------------------------------
// LayerNorm (one wave per token).  mode 0: window-partition row remap (src =
// original order, dst = window order).  mode 1: identity rows.
// ---------------------------------------------------------------------------
__global__ void ln_kernel(const float* __restrict__ x, const float* __restrict__ g,
                          const float* __restrict__ bvec, __bf16* __restrict__ out,
                          int mode) {
    int wave = (blockIdx.x * blockDim.x + threadIdx.x) >> 5;
    int lane = threadIdx.x & 31;
    if (wave >= TOK) return;
    int srow;
    if (mode == 0) {
        int w = wave / 49, i = wave % 49;
        int b = w >> 6, rem = w & 63, wh = rem >> 3, ww = rem & 7;
        int r = i / 7, c = i - r * 7;
        srow = b * 3136 + (wh * 7 + r) * 56 + (ww * 7 + c);
    } else {
        srow = wave;
    }
    const float* px = x + (size_t)srow * DIMC;
    float v[12];
    float s = 0.f;
#pragma unroll
    for (int j = 0; j < 12; ++j) { v[j] = px[lane + 32 * j]; s += v[j]; }
#pragma unroll
    for (int off = 16; off >= 1; off >>= 1) s += __shfl_xor(s, off, 32);
    float mean = s * (1.f / DIMC);
    float vs = 0.f;
#pragma unroll
    for (int j = 0; j < 12; ++j) { float d = v[j] - mean; vs += d * d; }
#pragma unroll
    for (int off = 16; off >= 1; off >>= 1) vs += __shfl_xor(vs, off, 32);
    float rstd = rsqrtf(vs * (1.f / DIMC) + 1e-5f);
    __bf16* po = out + (size_t)wave * DIMC;
#pragma unroll
    for (int j = 0; j < 12; ++j) {
        int c = lane + 32 * j;
        po[c] = (__bf16)((v[j] - mean) * rstd * g[c] + bvec[c]);
    }
}

// ---------------------------------------------------------------------------
// Generic WMMA GEMM: C[M][N] = A[M][K](bf16) x Bpacked + bias, per-wave 32x32.
// mode 0: out bf16, v=(acc+bias)*alpha
// mode 1: out fp32, v=acc+bias + resid[row]          (fc2 + residual)
// mode 2: out bf16, v=gelu(acc+bias)                 (fc1)
// mode 3: out fp32, window->orig row remap, + resid  (proj + residual x)
// ---------------------------------------------------------------------------
__global__ __launch_bounds__(256)
void gemm_wmma_kernel(const __bf16* __restrict__ A, const __bf16* __restrict__ Bp,
                      const float* __restrict__ bias, float alpha,
                      int M, int N, int K, int mode,
                      const float* __restrict__ resid,
                      float* __restrict__ outF, __bf16* __restrict__ outB) {
    int lane = threadIdx.x & 31;
    int wid  = (blockIdx.x * blockDim.x + threadIdx.x) >> 5;
    int nt32 = N >> 5;
    int mt = wid / nt32, nt = wid % nt32;
    int m0 = mt << 5, n0 = nt << 5;
    if (m0 >= M) return;
    int KT = K >> 5;

    v8f acc00 = zero_acc(), acc01 = zero_acc(), acc10 = zero_acc(), acc11 = zero_acc();

    // incrementing pointers: A rows advance 32 bf16 per K-step, packed B tiles
    // are contiguous (32*16 bf16 = 1024 B per kt step)
    int ahalf = (lane >= 16) ? 16 : 0;
    const __bf16* pa0 = A + (size_t)(m0 + (lane & 15)) * K + ahalf;
    const __bf16* pa1 = pa0 + (size_t)16 * K;
    const __bf16* pb0 = Bp + (((size_t)(n0 >> 4) * KT) * 32 + lane) * 16;
    const __bf16* pb1 = pb0 + (size_t)KT * 32 * 16;

    for (int kt = 0; kt < KT; ++kt) {
        v16bf a0 = load_a_frag_ptr(pa0);
        v16bf a1 = load_a_frag_ptr(pa1);
        v16bf b0 = load_b_frag_ptr(pb0);
        v16bf b1 = load_b_frag_ptr(pb1);
        acc00 = wmma_bf16(a0, b0, acc00);
        acc01 = wmma_bf16(a0, b1, acc01);
        acc10 = wmma_bf16(a1, b0, acc10);
        acc11 = wmma_bf16(a1, b1, acc11);
        pa0 += 32;
        pa1 += 32;
        pb0 += 32 * 16;
        pb1 += 32 * 16;
    }

    int lrow = (lane >= 16) ? 8 : 0;
    int lcol = lane & 15;
    v8f accs[4] = {acc00, acc01, acc10, acc11};
#pragma unroll
    for (int mi = 0; mi < 2; ++mi) {
#pragma unroll
        for (int ni = 0; ni < 2; ++ni) {
            v8f a = accs[mi * 2 + ni];
            int col = n0 + ni * 16 + lcol;
            float bv = bias ? bias[col] : 0.f;
#pragma unroll
            for (int r = 0; r < 8; ++r) {
                int row = m0 + mi * 16 + r + lrow;
                float v = (a[r] + bv) * alpha;
                if (mode == 2) {
                    v = 0.5f * v * (1.f + erff(v * 0.70710678118654752f));
                }
                if (mode == 0 || mode == 2) {
                    outB[(size_t)row * N + col] = (__bf16)v;
                } else if (mode == 1) {
                    size_t o = (size_t)row * N + col;
                    outF[o] = v + resid[o];
                } else {  // mode 3: window-order row -> original (B,L,C) row
                    int w = row / 49, i = row - w * 49;
                    int b = w >> 6, rem = w & 63, wh = rem >> 3, ww = rem & 7;
                    int rr = i / 7, cc = i - rr * 7;
                    size_t orow = (size_t)b * 3136 + (wh * 7 + rr) * 56 + ww * 7 + cc;
                    size_t o = orow * N + col;
                    outF[o] = v + resid[o];
                }
            }
        }
    }
}

// ---------------------------------------------------------------------------
// Windowed attention: one block per (window, head). 4 waves, each owns a
// 16-row strip of the padded 64x64 score matrix.
// ---------------------------------------------------------------------------
__global__ __launch_bounds__(128)
void attn_kernel(const __bf16* __restrict__ Q, const __bf16* __restrict__ Kb,
                 const __bf16* __restrict__ Vt, const float* __restrict__ table,
                 const int* __restrict__ rel, __bf16* __restrict__ O) {
    __shared__ __align__(16) float  S[64 * 64];
    __shared__ __align__(16) __bf16 P[64 * 64];

    int h = blockIdx.x % NHEAD;
    int w = blockIdx.x / NHEAD;
    int b = w >> 6;
    int lane = threadIdx.x & 31;
    int m = threadIdx.x >> 5;           // wave index 0..3 -> row strip
    int lrow = (lane >= 16) ? 8 : 0;
    int lcol = lane & 15;

    // ---- scores = (Q*scale) @ K^T  (Q already scaled), K=HD=32: 1 wmma/tile
    const __bf16* qbase = Q + (size_t)w * 49 * DIMC + h * HD;
    int arow = m * 16 + (lane & 15);
    v16bf aq = (arow < 49) ? load_a_frag(qbase, arow, DIMC, 0, lane)
                           : zero_frag16();
    const __bf16* kbase = Kb + ((size_t)(b * NHEAD + h)) * 49 * HD;
    v8f sc[4];
#pragma unroll
    for (int nt = 0; nt < 4; ++nt) {
        int c = nt * 16 + lcol;
        v16bf bk = (c < 49) ? load_b_frag_rowmajor(kbase, c, HD, 0, lane)
                            : zero_frag16();
        sc[nt] = wmma_bf16(aq, bk, zero_acc());
    }
    // bias + mask into LDS
#pragma unroll
    for (int nt = 0; nt < 4; ++nt) {
#pragma unroll
        for (int r = 0; r < 8; ++r) {
            int i = m * 16 + r + lrow;
            int j = nt * 16 + lcol;
            float v = sc[nt][r];
            if (j < 49) {
                if (i < 49) v += table[rel[i * 49 + j] * NHEAD + h];
            } else {
                v = -1e30f;
            }
            S[i * 64 + j] = v;
        }
    }
    __syncthreads();

    // ---- softmax over each row of the strip (cols 0..48 live, rest -inf)
    for (int r = 0; r < 16; ++r) {
        int i = m * 16 + r;
        float v0 = S[i * 64 + lane];
        float v1 = S[i * 64 + 32 + lane];
        float mx = fmaxf(v0, v1);
#pragma unroll
        for (int off = 16; off >= 1; off >>= 1) mx = fmaxf(mx, __shfl_xor(mx, off, 32));
        float e0 = __expf(v0 - mx);
        float e1 = __expf(v1 - mx);
        float s = e0 + e1;
#pragma unroll
        for (int off = 16; off >= 1; off >>= 1) s += __shfl_xor(s, off, 32);
        float inv = 1.f / s;
        P[i * 64 + lane]      = (__bf16)(e0 * inv);
        P[i * 64 + 32 + lane] = (__bf16)(e1 * inv);
    }
    __syncthreads();

    // ---- O strip = P(16x64) @ V(64x32), K=64 -> 2 wmma per 16-col tile
    const __bf16* vbase = Vt + ((size_t)(b * NHEAD + h)) * HD * 64;
#pragma unroll
    for (int nt2 = 0; nt2 < 2; ++nt2) {
        v8f ao = zero_acc();
#pragma unroll
        for (int ks = 0; ks < 2; ++ks) {
            int k0 = ks * 32;
            v16bf ap = load_a_frag(P, m * 16 + (lane & 15), 64, k0, lane);
            v16bf bv = load_b_frag_rowmajor(vbase, nt2 * 16 + lcol, 64, k0, lane);
            ao = wmma_bf16(ap, bv, ao);
        }
#pragma unroll
        for (int r = 0; r < 8; ++r) {
            int i = m * 16 + r + lrow;
            if (i < 49)
                O[((size_t)w * 49 + i) * DIMC + h * HD + nt2 * 16 + lcol] = (__bf16)ao[r];
        }
    }
}

// ---------------------------------------------------------------------------
// Host-side orchestration
// ---------------------------------------------------------------------------
extern "C" void kernel_launch(void* const* d_in, const int* in_sizes, int n_in,
                              void* d_out, int out_size, void* d_ws, size_t ws_size,
                              hipStream_t stream) {
    (void)in_sizes; (void)n_in; (void)out_size; (void)ws_size;
    const float* x        = (const float*)d_in[0];
    const float* kv       = (const float*)d_in[1];
    const int*   rel      = (const int*)  d_in[2];
    const float* g1       = (const float*)d_in[5];
    const float* b1       = (const float*)d_in[6];
    const float* Wqkv     = (const float*)d_in[7];
    const float* bqkv     = (const float*)d_in[8];
    const float* btable   = (const float*)d_in[9];
    const float* Wproj    = (const float*)d_in[10];
    const float* bproj    = (const float*)d_in[11];
    const float* g2       = (const float*)d_in[12];
    const float* b2       = (const float*)d_in[13];
    const float* Wfc1     = (const float*)d_in[14];
    const float* bfc1     = (const float*)d_in[15];
    const float* Wfc2     = (const float*)d_in[16];
    const float* bfc2     = (const float*)d_in[17];
    float* out = (float*)d_out;

    // ---- workspace carving (256-byte aligned)
    char* p = (char*)d_ws;
    auto carve = [&](size_t bytes) {
        void* r = (void*)p;
        p += (bytes + 255) & ~(size_t)255;
        return r;
    };
    const size_t ACT2  = (size_t)TOK * DIMC * 2;   // bf16 activation
    __bf16* A1     = (__bf16*)carve(ACT2);                 // LN1 out / reused as LN2 out
    __bf16* Qb     = (__bf16*)carve(ACT2);
    __bf16* Ob     = (__bf16*)carve(ACT2);
    float*  R1     = (float*) carve((size_t)TOK * DIMC * 4);
    __bf16* G      = (__bf16*)carve((size_t)TOK * HID * 2);
    __bf16* kbf    = (__bf16*)carve((size_t)16 * NHEAD * 49 * HD * 2);
    __bf16* vT     = (__bf16*)carve((size_t)16 * NHEAD * HD * 64 * 2);
    __bf16* Wq_p   = (__bf16*)carve((size_t)DIMC * DIMC * 2);
    __bf16* Wpr_p  = (__bf16*)carve((size_t)DIMC * DIMC * 2);
    __bf16* Wf1_p  = (__bf16*)carve((size_t)DIMC * HID * 2);
    __bf16* Wf2_p  = (__bf16*)carve((size_t)HID * DIMC * 2);

    // ---- weight packing
    packw_kernel<<<36, 256, 0, stream>>>(Wqkv, 3 * DIMC, 0, DIMC, DIMC, Wq_p);
    packw_kernel<<<36, 256, 0, stream>>>(Wproj, DIMC, 0, DIMC, DIMC, Wpr_p);
    packw_kernel<<<144, 256, 0, stream>>>(Wfc1, HID, 0, DIMC, HID, Wf1_p);
    packw_kernel<<<144, 256, 0, stream>>>(Wfc2, DIMC, 0, HID, DIMC, Wf2_p);

    // ---- kv -> bf16 (k row-major, v transposed + zero padded)
    kvprep_kernel<<<1536, 256, 0, stream>>>(kv, kbf, vT);

    // ---- LN1 with window partition
    ln_kernel<<<6272, 256, 0, stream>>>(x, g1, b1, A1, 0);

    // ---- Q = (LN1 @ Wq + bq) * scale   [50176x384x384]
    gemm_wmma_kernel<<<2352, 256, 0, stream>>>(A1, Wq_p, bqkv, QSCALE,
                                               TOK, DIMC, DIMC, 0, nullptr,
                                               nullptr, Qb);

    // ---- windowed attention
    attn_kernel<<<NWIN * NHEAD, 128, 0, stream>>>(Qb, kbf, vT, btable, rel, Ob);

    // ---- proj + window_reverse + residual: R1 = x + (O @ Wproj + bproj)
    gemm_wmma_kernel<<<2352, 256, 0, stream>>>(Ob, Wpr_p, bproj, 1.f,
                                               TOK, DIMC, DIMC, 3, x,
                                               R1, nullptr);

    // ---- LN2 (identity row order), reuse A1
    ln_kernel<<<6272, 256, 0, stream>>>(R1, g2, b2, A1, 1);

    // ---- fc1 + exact GELU  [50176x384x1536]
    gemm_wmma_kernel<<<9408, 256, 0, stream>>>(A1, Wf1_p, bfc1, 1.f,
                                               TOK, HID, DIMC, 2, nullptr,
                                               nullptr, G);

    // ---- fc2 + residual -> output  [50176x1536x384]
    gemm_wmma_kernel<<<2352, 256, 0, stream>>>(G, Wf2_p, bfc2, 1.f,
                                               TOK, DIMC, HID, 1, R1,
                                               out, nullptr);
}
